// DGCNNet_9852654977191
// MI455X (gfx1250) — matrix operand order repaired and checked by reading, not compiled, not run on hardware
//
#include <hip/hip_runtime.h>
#include <cstdint>

// ---------- types ----------
typedef __bf16 bf16;
typedef __attribute__((ext_vector_type(16))) __bf16 v16bf;
typedef __attribute__((ext_vector_type(8)))  float  v8f;

#define BATCH 32
#define PTS   1024
#define KNN   20

// ---------- WMMA helpers (CDNA5 16x16x32 bf16, wave32) ----------
__device__ inline v8f bfwmma(v16bf a, v16bf b, v8f c) {
  return __builtin_amdgcn_wmma_f32_16x16x32_bf16(false, a, false, b, (short)0, c,
                                                 false, false);
}

// Async global->LDS DMA (CDNA5). GVS mode: uniform 64-bit SGPR base,
// per-lane 32-bit VGPR byte offset, per-lane LDS byte address in VDST.
// Tracked by ASYNCcnt; wait with s_wait_asynccnt.
__device__ inline void async_load_b128(uint32_t lds_byte, uint32_t voff,
                                       const void* base) {
  asm volatile("global_load_async_to_lds_b128 %0, %1, %2"
               :: "v"(lds_byte), "v"(voff), "s"(base)
               : "memory");
}
__device__ inline void wait_async0() {
  asm volatile("s_wait_asynccnt 0x0" ::: "memory");
}

// A fragment: 16x32 (M x K), src row-major with leading dim ld (bf16 units).
// Layout: lanes 0-15 hold M=0..15, K 0..15 contiguous;
// lanes 16-31 hold K 16..31 contiguous.
__device__ inline v16bf load_frag_A(const bf16* p0, int ld) {
  const int lane = threadIdx.x & 31;
  const bf16* p = p0 + (lane & 15) * ld + ((lane >> 4) << 4);
  v16bf a;
#pragma unroll
  for (int v = 0; v < 4; ++v) { a[2 * v] = p[2 * v]; a[2 * v + 1] = p[2 * v + 1]; }
#pragma unroll
  for (int v = 4; v < 8; ++v) {
    a[2 * v]     = p[8 + 2 * (v - 4)];
    a[2 * v + 1] = p[8 + 2 * (v - 4) + 1];
  }
  return a;
}

// B fragment: 32x16 (K x N). Source is "N-major": 16 rows of N, each holding
// K contiguously with leading dim ld. Lane n = lane&15 picks the row, reads
// 16 contiguous K values at khalf = (lane>>4)*16.
__device__ inline v16bf load_frag_B(const bf16* p0, int ld) {
  const int lane = threadIdx.x & 31;
  const bf16* p = p0 + (lane & 15) * ld + ((lane >> 4) << 4);
  v16bf b;
#pragma unroll
  for (int e = 0; e < 16; ++e) b[e] = p[e];
  return b;
}

// ---------- top-k (register-resident, unrolled insertion) ----------
__device__ inline void topk_init(float* bd, int* bi) {
#pragma unroll
  for (int s = 0; s < KNN; ++s) { bd[s] = 3.0e38f; bi[s] = 0; }
}
__device__ inline void topk_insert(float* bd, int* bi, float d, int j) {
  if (d >= bd[KNN - 1]) return;
#pragma unroll
  for (int s = KNN - 1; s > 0; --s) {
    if (d < bd[s - 1]) { bd[s] = bd[s - 1]; bi[s] = bi[s - 1]; }
    else if (d < bd[s]) { bd[s] = d; bi[s] = j; }
  }
  if (d < bd[0]) { bd[0] = d; bi[0] = j; }
}

// ---------- generic WMMA MLP layer over an LDS edge tile ----------
// inbuf [rows][ldin] bf16 (LDS), outbuf [rows][ldout] bf16 (LDS),
// WT [cout][cin] bf16 (global, pre-transposed). epilogue: relu(z+b)*sc+sh -> bf16
__device__ inline void mlp_layer_tile(const bf16* inbuf, int ldin,
                                      bf16* outbuf, int ldout,
                                      const bf16* __restrict__ WT, int cin, int cout,
                                      const float* __restrict__ bias,
                                      const float* __restrict__ bn,
                                      int rows) {
  const int tid = threadIdx.x;
  const int wave = tid >> 5, lane = tid & 31;
  const int nwaves = blockDim.x >> 5;
  const int mtiles = rows >> 4, ntiles = cout >> 4, ksteps = cin >> 5;
  for (int job = wave; job < mtiles * ntiles; job += nwaves) {
    const int mt = job / ntiles, nt = job - mt * ntiles;
    v8f acc = {};
    for (int kk = 0; kk < ksteps; ++kk) {
      v16bf a = load_frag_A(inbuf + (mt * 16) * ldin + kk * 32, ldin);
      v16bf b = load_frag_B(WT + (size_t)(nt * 16) * cin + kk * 32, cin);
      acc = bfwmma(a, b, acc);
    }
    const int col = nt * 16 + (lane & 15);
    const float sc = bn[col], sh = bn[cout + col], bs = bias[col];
    const int rbase = mt * 16 + ((lane >> 4) << 3);
#pragma unroll
    for (int v = 0; v < 8; ++v) {
      float u = acc[v] + bs;
      u = u > 0.f ? u : 0.f;
      u = __builtin_fmaf(sc, u, sh);
      outbuf[(rbase + v) * ldout + col] = (bf16)u;
    }
  }
}

// ---------- K0: weight convert/transpose + BN fold ----------
struct Params {
  const float* W[8];   // c1a c1b c1c c2 l1 m1 m2 fin
  const float* b[8];
  const float* g[7];   // bn layers: c1a c1b c1c c2 l1 m1 m2
  const float* bt[7];
  const float* m[7];
  const float* v[7];
};

__global__ void prep_kernel(Params P, bf16* W1aT, bf16* W1bT, bf16* W1cT,
                            bf16* W2T, bf16* Wl1T,
                            float* bn_c1a, float* bn_c1b, float* bn_c1c,
                            float* bn_c2, float* bn_l1, float* bn_m1, float* bn_m2) {
  const int gid = blockIdx.x * blockDim.x + threadIdx.x;
  const int stride = gridDim.x * blockDim.x;
  for (int i = gid; i < 64 * 32; i += stride) {       // 6->64, K padded to 32
    const int n = i >> 5, k = i & 31;
    W1aT[i] = (bf16)(k < 6 ? P.W[0][k * 64 + n] : 0.f);
  }
  for (int i = gid; i < 64 * 64; i += stride) {
    const int n = i >> 6, k = i & 63;
    W1bT[i] = (bf16)P.W[1][k * 64 + n];
    W1cT[i] = (bf16)P.W[2][k * 64 + n];
  }
  for (int i = gid; i < 128 * 128; i += stride) {
    const int n = i >> 7, k = i & 127;
    W2T[i] = (bf16)P.W[3][k * 128 + n];
  }
  for (int i = gid; i < 1024 * 192; i += stride) {
    const int n = i / 192, k = i - n * 192;
    Wl1T[i] = (bf16)P.W[4][(size_t)k * 1024 + n];
  }
  auto dobn = [&](int L, int C, float* out) {
    for (int c = gid; c < C; c += stride) {
      const float sc = P.g[L][c] * rsqrtf(P.v[L][c] + 1e-5f);
      out[c] = sc;
      out[C + c] = P.bt[L][c] - P.m[L][c] * sc;
    }
  };
  dobn(0, 64, bn_c1a); dobn(1, 64, bn_c1b); dobn(2, 64, bn_c1c);
  dobn(3, 128, bn_c2); dobn(4, 1024, bn_l1);
  dobn(5, 512, bn_m1); dobn(6, 256, bn_m2);
}

// ---------- K1: kNN on positions (C=3) ----------
__global__ __launch_bounds__(256) void knn_pos_kernel(const float* __restrict__ pos,
                                                      int* __restrict__ idx1) {
  __shared__ float px[PTS], py[PTS], pz[PTS];
  const int cloud = blockIdx.x >> 2;
  const int base = (blockIdx.x & 3) << 8;
  const int t = threadIdx.x;
  for (int i = t; i < PTS; i += 256) {
    const float* p = pos + (size_t)(cloud * PTS + i) * 3;
    px[i] = p[0]; py[i] = p[1]; pz[i] = p[2];
  }
  __syncthreads();
  const int i = base + t;
  const float xi = px[i], yi = py[i], zi = pz[i];
  float bd[KNN]; int bi[KNN];
  topk_init(bd, bi);
  for (int j = 0; j < PTS; ++j) {
    const float dx = px[j] - xi, dy = py[j] - yi, dz = pz[j] - zi;
    topk_insert(bd, bi, dx * dx + dy * dy + dz * dz, j);
  }
  int* o = idx1 + (size_t)(cloud * PTS + i) * KNN;
#pragma unroll
  for (int s = 0; s < KNN; ++s) o[s] = bi[s];
}

// ---------- K2: edgeconv1 (6->64->64->64, max over k) ----------
__global__ __launch_bounds__(128) void edgeconv1_kernel(
    const float* __restrict__ pos, const int* __restrict__ idx1,
    const bf16* __restrict__ W1aT, const bf16* __restrict__ W1bT,
    const bf16* __restrict__ W1cT,
    const float* __restrict__ b1a, const float* __restrict__ b1b,
    const float* __restrict__ b1c,
    const float* __restrict__ bn1a, const float* __restrict__ bn1b,
    const float* __restrict__ bn1c,
    bf16* __restrict__ x1bf) {
  __shared__ alignas(16) bf16 bufA[160 * 32];   // edge feats, K padded 6->32
  __shared__ alignas(16) bf16 bufB[160 * 64];
  __shared__ alignas(16) bf16 bufC[160 * 64];
  const int tid = threadIdx.x;
  const int pbase = blockIdx.x * 8;
  const int cloud = pbase >> 10;
  for (int t = tid; t < 160 * 32; t += 128) {
    const int e = t >> 5, c = t & 31;
    const int pl = e / KNN, kk = e - pl * KNN;
    const int gi = pbase + pl;
    const int gj = (cloud << 10) + idx1[(size_t)gi * KNN + kk];
    float val = 0.f;
    if (c < 3)      val = pos[(size_t)gi * 3 + c];
    else if (c < 6) val = pos[(size_t)gj * 3 + (c - 3)] - pos[(size_t)gi * 3 + (c - 3)];
    bufA[t] = (bf16)val;
  }
  __syncthreads();
  mlp_layer_tile(bufA, 32, bufB, 64, W1aT, 32, 64, b1a, bn1a, 160);
  __syncthreads();
  mlp_layer_tile(bufB, 64, bufC, 64, W1bT, 64, 64, b1b, bn1b, 160);
  __syncthreads();
  mlp_layer_tile(bufC, 64, bufB, 64, W1cT, 64, 64, b1c, bn1c, 160);
  __syncthreads();
  for (int t = tid; t < 8 * 64; t += 128) {
    const int pl = t >> 6, c = t & 63;
    float mx = -3.0e38f;
#pragma unroll
    for (int r = 0; r < KNN; ++r) mx = fmaxf(mx, (float)bufB[(pl * KNN + r) * 64 + c]);
    x1bf[(size_t)(pbase + pl) * 64 + c] = (bf16)mx;
  }
}

// ---------- K3: kNN in feature space via WMMA Gram (rank by |xj|^2 - 2 xi.xj) ----------
__global__ __launch_bounds__(128) void knn_feat_kernel(const bf16* __restrict__ x1bf,
                                                       int* __restrict__ idx2) {
  __shared__ alignas(16) bf16 Bch[64 * 64];   // 64 candidate rows x 64 feats
  __shared__ float sqj[64];
  __shared__ float dch[4][16][64];            // per-wave distance chunk
  __shared__ float mE[4][16][40];
  __shared__ int   mI[4][16][40];
  const int tid = threadIdx.x, wave = tid >> 5, lane = tid & 31;
  const int cloud = blockIdx.x >> 4;
  const int rbase = ((blockIdx.x & 15) << 6) + (wave << 4);  // wave's 16 rows (cloud-local)
  const bf16* X = x1bf + ((size_t)cloud << 10) * 64;
  const v16bf a0 = load_frag_A(X + (size_t)rbase * 64, 64);
  const v16bf a1 = load_frag_A(X + (size_t)rbase * 64 + 32, 64);
  const int r = lane & 15, h = lane >> 4;
  const uint32_t BchLds = (uint32_t)(uintptr_t)(void*)Bch;
  float bd[KNN]; int bi[KNN];
  topk_init(bd, bi);
  for (int ch = 0; ch < 16; ++ch) {
    __syncthreads();
    { // async DMA: candidate block (rows ch*64 .. +64) straight into LDS
      const void* srcbase = (const void*)(X + (size_t)(ch << 6) * 64);
      for (int t2 = tid; t2 < 512; t2 += 128)              // 512 x 16B = 8KB
        async_load_b128(BchLds + (uint32_t)(t2 * 16), (uint32_t)(t2 * 16), srcbase);
      wait_async0();
    }
    __syncthreads();
    if (tid < 64) {
      float s = 0.f;
      for (int k = 0; k < 64; ++k) { const float x = (float)Bch[tid * 64 + k]; s += x * x; }
      sqj[tid] = s;
    }
    __syncthreads();
#pragma unroll
    for (int jt = 0; jt < 4; ++jt) {
      v8f acc = {};
      acc = bfwmma(a0, load_frag_B(Bch + (jt * 16) * 64, 64), acc);
      acc = bfwmma(a1, load_frag_B(Bch + (jt * 16) * 64 + 32, 64), acc);
      const int jloc = jt * 16 + (lane & 15);
      const int rb = (lane >> 4) << 3;
      const float sq = sqj[jloc];
#pragma unroll
      for (int v = 0; v < 8; ++v) dch[wave][rb + v][jloc] = sq - 2.f * acc[v];
    }
    // per-wave selection: lane owns (row r, column half h); wave-internal LDS
    for (int cc = 0; cc < 32; ++cc) {
      const int col = (h << 5) + cc;
      topk_insert(bd, bi, dch[wave][r][col], (ch << 6) + col);
    }
  }
#pragma unroll
  for (int s = 0; s < KNN; ++s) { mE[wave][r][h * KNN + s] = bd[s]; mI[wave][r][h * KNN + s] = bi[s]; }
  __syncthreads();
  if (lane < 16) {
    float fd[KNN]; int fi[KNN];
    topk_init(fd, fi);
    for (int s = 0; s < 2 * KNN; ++s) topk_insert(fd, fi, mE[wave][lane][s], mI[wave][lane][s]);
    int* o = idx2 + ((size_t)(cloud << 10) + rbase + lane) * KNN;
#pragma unroll
    for (int s = 0; s < KNN; ++s) o[s] = fi[s];
  }
}

// ---------- K4: edgeconv2 (128->128, max over k) ----------
__global__ __launch_bounds__(128) void edgeconv2_kernel(
    const bf16* __restrict__ x1bf, const int* __restrict__ idx2,
    const bf16* __restrict__ W2T, const float* __restrict__ b2,
    const float* __restrict__ bn2, bf16* __restrict__ x2bf) {
  __shared__ alignas(16) uint32_t bufA32[80 * 64];   // [80 edges][128 bf16]
  __shared__ alignas(16) bf16 bufB[80 * 128];
  bf16* bufA = (bf16*)bufA32;
  const int tid = threadIdx.x;
  const int pbase = blockIdx.x * 4;
  const int cloud = pbase >> 10;
  // x_i half (cols 0..63): pure row replication -> async DMA, 16B chunks
  {
    const uint32_t lds0 = (uint32_t)(uintptr_t)(void*)bufA32;
    for (int t = tid; t < 80 * 8; t += 128) {
      const int e = t >> 3, q = t & 7;
      const int gi = pbase + e / KNN;
      async_load_b128(lds0 + (uint32_t)(e * 256 + q * 16),
                      (uint32_t)(gi * 128 + q * 16), x1bf);
    }
  }
  // (x_j - x_i) half (cols 64..127): dword-paired loads + repack
  union bfpair { uint32_t u; bf16 h[2]; };
  for (int t = tid; t < 80 * 32; t += 128) {
    const int e = t >> 5, c2 = t & 31;
    const int pl = e / KNN, kk = e - pl * KNN;
    const int gi = pbase + pl;
    const int gj = (cloud << 10) + idx2[(size_t)gi * KNN + kk];
    bfpair pi, pj, po;
    pi.u = ((const uint32_t*)(x1bf + (size_t)gi * 64))[c2];
    pj.u = ((const uint32_t*)(x1bf + (size_t)gj * 64))[c2];
    po.h[0] = (bf16)((float)pj.h[0] - (float)pi.h[0]);
    po.h[1] = (bf16)((float)pj.h[1] - (float)pi.h[1]);
    bufA32[e * 64 + 32 + c2] = po.u;
  }
  wait_async0();
  __syncthreads();
  mlp_layer_tile(bufA, 128, bufB, 128, W2T, 128, 128, b2, bn2, 80);
  __syncthreads();
  for (int t = tid; t < 4 * 128; t += 128) {
    const int pl = t >> 7, c = t & 127;
    float mx = -3.0e38f;
#pragma unroll
    for (int rr = 0; rr < KNN; ++rr) mx = fmaxf(mx, (float)bufB[(pl * KNN + rr) * 128 + c]);
    x2bf[(size_t)(pbase + pl) * 128 + c] = (bf16)mx;
  }
}

// ---------- K5: lin1 (192->1024) fused with global max pool ----------
__global__ __launch_bounds__(128) void lin1_pool_kernel(
    const bf16* __restrict__ x1bf, const bf16* __restrict__ x2bf,
    const bf16* __restrict__ Wl1T, const float* __restrict__ bl1,
    const float* __restrict__ bnl1, float* __restrict__ pooled) {
  __shared__ float lmax[4][32];
  const int tid = threadIdx.x, wave = tid >> 5, lane = tid & 31;
  const int cloud = blockIdx.x >> 4;
  const int ncol = ((blockIdx.x & 15) << 6) + (wave << 4);  // wave's 16 output cols
  v16bf bfr[6];
#pragma unroll
  for (int kk = 0; kk < 6; ++kk)
    bfr[kk] = load_frag_B(Wl1T + (size_t)ncol * 192 + kk * 32, 192);
  float rmax[8];
#pragma unroll
  for (int v = 0; v < 8; ++v) rmax[v] = -3.0e38f;
  const int col = ncol + (lane & 15);
  const float sc = bnl1[col], sh = bnl1[1024 + col], bs = bl1[col];
  for (int mt = 0; mt < 64; ++mt) {
    const size_t row0 = ((size_t)cloud << 10) + ((size_t)mt << 4);
    if (mt + 1 < 64) {
      __builtin_prefetch(x1bf + (row0 + 16) * 64, 0, 1);
      __builtin_prefetch(x2bf + (row0 + 16) * 128, 0, 1);
    }
    v8f acc = {};
    acc = bfwmma(load_frag_A(x1bf + row0 * 64, 64),        bfr[0], acc);
    acc = bfwmma(load_frag_A(x1bf + row0 * 64 + 32, 64),   bfr[1], acc);
    acc = bfwmma(load_frag_A(x2bf + row0 * 128, 128),      bfr[2], acc);
    acc = bfwmma(load_frag_A(x2bf + row0 * 128 + 32, 128), bfr[3], acc);
    acc = bfwmma(load_frag_A(x2bf + row0 * 128 + 64, 128), bfr[4], acc);
    acc = bfwmma(load_frag_A(x2bf + row0 * 128 + 96, 128), bfr[5], acc);
#pragma unroll
    for (int v = 0; v < 8; ++v) {
      float u = acc[v] + bs;
      u = u > 0.f ? u : 0.f;
      u = __builtin_fmaf(sc, u, sh);
      rmax[v] = fmaxf(rmax[v], u);
    }
  }
  float m = rmax[0];
#pragma unroll
  for (int v = 1; v < 8; ++v) m = fmaxf(m, rmax[v]);
  lmax[wave][lane] = m;
  __syncthreads();
  if (lane < 16)
    pooled[((size_t)cloud << 10) + ncol + lane] = fmaxf(lmax[wave][lane], lmax[wave][lane + 16]);
}

// ---------- K6: head MLPs (small; scalar f32) ----------
__global__ void head_mlp_kernel(const float* __restrict__ in, const float* __restrict__ W,
                                const float* __restrict__ bias, const float* __restrict__ bn,
                                float* __restrict__ out, int cin, int cout) {
  const int gid = blockIdx.x * blockDim.x + threadIdx.x;
  if (gid >= BATCH * cout) return;
  const int b = gid / cout, c = gid - b * cout;
  const float* row = in + (size_t)b * cin;
  float s = 0.f;
  for (int k = 0; k < cin; ++k) s = __builtin_fmaf(row[k], W[(size_t)k * cout + c], s);
  s += bias[c];
  s = s > 0.f ? s : 0.f;
  out[gid] = __builtin_fmaf(bn[c], s, bn[cout + c]);
}

__global__ __launch_bounds__(64) void final_kernel(const float* __restrict__ h2,
                                                   const float* __restrict__ Wf,
                                                   const float* __restrict__ bfin,
                                                   float* __restrict__ out) {
  __shared__ float z[40];
  __shared__ float red[2];
  const int b = blockIdx.x, t = threadIdx.x;
  if (t < 40) {
    const float* row = h2 + (size_t)b * 256;
    float s = bfin[t];
    for (int k = 0; k < 256; ++k) s = __builtin_fmaf(row[k], Wf[k * 40 + t], s);
    z[t] = s;
  }
  __syncthreads();
  if (t == 0) {
    float mx = -3.0e38f;
    for (int c = 0; c < 40; ++c) mx = fmaxf(mx, z[c]);
    float se = 0.f;
    for (int c = 0; c < 40; ++c) se += __expf(z[c] - mx);
    red[0] = mx; red[1] = __logf(se);
  }
  __syncthreads();
  if (t < 40) out[b * 40 + t] = z[t] - red[0] - red[1];
}

// ---------- launch ----------
extern "C" void kernel_launch(void* const* d_in, const int* in_sizes, int n_in,
                              void* d_out, int out_size, void* d_ws, size_t ws_size,
                              hipStream_t stream) {
  (void)in_sizes; (void)n_in; (void)out_size; (void)ws_size;
  const float* pos = (const float*)d_in[0];
  Params P;
  const int wi[8] = {2, 8, 14, 20, 26, 32, 38, 44};
  for (int l = 0; l < 8; ++l) {
    P.W[l] = (const float*)d_in[wi[l]];
    P.b[l] = (const float*)d_in[wi[l] + 1];
  }
  const int bni[7] = {4, 10, 16, 22, 28, 34, 40};
  for (int l = 0; l < 7; ++l) {
    P.g[l]  = (const float*)d_in[bni[l]];
    P.bt[l] = (const float*)d_in[bni[l] + 1];
    P.m[l]  = (const float*)d_in[bni[l] + 2];
    P.v[l]  = (const float*)d_in[bni[l] + 3];
  }

  size_t off = 0;
  auto alloc = [&](size_t bytes) -> void* {
    void* p = (char*)d_ws + off;
    off = (off + bytes + 255) & ~(size_t)255;
    return p;
  };
  bf16* W1aT = (bf16*)alloc(64 * 32 * sizeof(bf16));
  bf16* W1bT = (bf16*)alloc(64 * 64 * sizeof(bf16));
  bf16* W1cT = (bf16*)alloc(64 * 64 * sizeof(bf16));
  bf16* W2T  = (bf16*)alloc(128 * 128 * sizeof(bf16));
  bf16* Wl1T = (bf16*)alloc((size_t)1024 * 192 * sizeof(bf16));
  float* bn_c1a = (float*)alloc(2 * 64 * sizeof(float));
  float* bn_c1b = (float*)alloc(2 * 64 * sizeof(float));
  float* bn_c1c = (float*)alloc(2 * 64 * sizeof(float));
  float* bn_c2  = (float*)alloc(2 * 128 * sizeof(float));
  float* bn_l1  = (float*)alloc(2 * 1024 * sizeof(float));
  float* bn_m1  = (float*)alloc(2 * 512 * sizeof(float));
  float* bn_m2  = (float*)alloc(2 * 256 * sizeof(float));
  int*  idx1 = (int*)alloc((size_t)BATCH * PTS * KNN * sizeof(int));
  int*  idx2 = (int*)alloc((size_t)BATCH * PTS * KNN * sizeof(int));
  bf16* x1bf = (bf16*)alloc((size_t)BATCH * PTS * 64 * sizeof(bf16));
  bf16* x2bf = (bf16*)alloc((size_t)BATCH * PTS * 128 * sizeof(bf16));
  float* pooled = (float*)alloc((size_t)BATCH * 1024 * sizeof(float));
  float* h1 = (float*)alloc((size_t)BATCH * 512 * sizeof(float));
  float* h2 = (float*)alloc((size_t)BATCH * 256 * sizeof(float));

  prep_kernel<<<512, 256, 0, stream>>>(P, W1aT, W1bT, W1cT, W2T, Wl1T,
                                       bn_c1a, bn_c1b, bn_c1c, bn_c2, bn_l1, bn_m1, bn_m2);
  knn_pos_kernel<<<BATCH * 4, 256, 0, stream>>>(pos, idx1);
  edgeconv1_kernel<<<BATCH * PTS / 8, 128, 0, stream>>>(
      pos, idx1, W1aT, W1bT, W1cT, P.b[0], P.b[1], P.b[2], bn_c1a, bn_c1b, bn_c1c, x1bf);
  knn_feat_kernel<<<BATCH * 16, 128, 0, stream>>>(x1bf, idx2);
  edgeconv2_kernel<<<BATCH * PTS / 4, 128, 0, stream>>>(x1bf, idx2, W2T, P.b[3], bn_c2, x2bf);
  lin1_pool_kernel<<<BATCH * 16, 128, 0, stream>>>(x1bf, x2bf, Wl1T, P.b[4], bn_l1, pooled);
  head_mlp_kernel<<<(BATCH * 512 + 255) / 256, 256, 0, stream>>>(pooled, P.W[5], P.b[5], bn_m1, h1, 1024, 512);
  head_mlp_kernel<<<(BATCH * 256 + 255) / 256, 256, 0, stream>>>(h1, P.W[6], P.b[6], bn_m2, h2, 512, 256);
  final_kernel<<<BATCH, 64, 0, stream>>>(h2, P.W[7], P.b[7], (float*)d_out);
}
